// AttentionLSTMDecoder_62620623176078
// MI455X (gfx1250) — compile-verified, matrix operand order checked by reading, not run on hardware
//
#include <hip/hip_runtime.h>
#include <hip/hip_bf16.h>
#include <math.h>

// ---------------------------------------------------------------------------
// AttentionLSTMDecoder for MI455X (gfx1250).
// B=64, T=20, H=512, E=512, V=32000, N=8192.
// Weights + node_features pre-converted to bf16 ONCE; recurrent h kept in
// bf16, c in f32. GEMM: each wave computes a 64x16 output slab (4 M-tiles,
// one N-tile) so every weight tile is loaded once per K-chunk and reused by
// 4 v_wmma_f32_16x16x32_bf16 ops; K-chunks are software-pipelined (register
// double-buffering) so loads stay in flight behind the WMMAs.
// ---------------------------------------------------------------------------

typedef __attribute__((ext_vector_type(16))) __bf16 vbf16x16;
typedef __attribute__((ext_vector_type(8)))  __bf16 vbf16x8;
typedef __attribute__((ext_vector_type(8)))  float  vf32x8;

__device__ __forceinline__ __bf16 f2bf(float f) {
    unsigned u = __builtin_bit_cast(unsigned, f);
    unsigned r = u + 0x7FFFu + ((u >> 16) & 1u);     // round-to-nearest-even
    return __builtin_bit_cast(__bf16, (unsigned short)(r >> 16));
}

// Load 16 bf16 K-values for one matrix row in the CDNA5 16-bit A/B layout.
// p points at row_base + kbase + (lane>=16 ? 8 : 0) in elements.
// elements 0..7  <- K = kbase + sel + 0..7     (VGPR0..3)
// elements 8..15 <- K = kbase + 16 + sel + 0..7 (VGPR4..7)
__device__ __forceinline__ vbf16x16 load_bf16_row(const __bf16* __restrict__ p) {
    vbf16x8 lo = *reinterpret_cast<const vbf16x8*>(p);        // 16B
    vbf16x8 hi = *reinterpret_cast<const vbf16x8*>(p + 16);   // 16B
    return __builtin_shufflevector(lo, hi, 0, 1, 2, 3, 4, 5, 6, 7,
                                           8, 9, 10, 11, 12, 13, 14, 15);
}

// Software-pipelined K-loop for one (A,W) pair, accumulating into acc[0..3].
// arow: A base for this wave's 64-row slab (row m0+r), wrow: W base (row n0+r).
__device__ __forceinline__ void gemm_k_block(const __bf16* __restrict__ arow, int lda,
                                             const __bf16* __restrict__ wrow, int K,
                                             vf32x8 acc[4])
{
    vbf16x16 bcur = load_bf16_row(wrow);
    vbf16x16 acur0 = load_bf16_row(arow);
    vbf16x16 acur1 = load_bf16_row(arow + (size_t)16 * lda);
    vbf16x16 acur2 = load_bf16_row(arow + (size_t)32 * lda);
    vbf16x16 acur3 = load_bf16_row(arow + (size_t)48 * lda);
    for (int k = 32; k < K; k += 32) {
        // prefetch next chunk before consuming current one
        vbf16x16 bnxt = load_bf16_row(wrow + k);
        vbf16x16 anxt0 = load_bf16_row(arow + k);
        vbf16x16 anxt1 = load_bf16_row(arow + (size_t)16 * lda + k);
        vbf16x16 anxt2 = load_bf16_row(arow + (size_t)32 * lda + k);
        vbf16x16 anxt3 = load_bf16_row(arow + (size_t)48 * lda + k);
        acc[0] = __builtin_amdgcn_wmma_f32_16x16x32_bf16(false, acur0, false, bcur,
                                                         (short)0, acc[0], false, false);
        acc[1] = __builtin_amdgcn_wmma_f32_16x16x32_bf16(false, acur1, false, bcur,
                                                         (short)0, acc[1], false, false);
        acc[2] = __builtin_amdgcn_wmma_f32_16x16x32_bf16(false, acur2, false, bcur,
                                                         (short)0, acc[2], false, false);
        acc[3] = __builtin_amdgcn_wmma_f32_16x16x32_bf16(false, acur3, false, bcur,
                                                         (short)0, acc[3], false, false);
        bcur = bnxt;
        acur0 = anxt0; acur1 = anxt1; acur2 = anxt2; acur3 = anxt3;
    }
    acc[0] = __builtin_amdgcn_wmma_f32_16x16x32_bf16(false, acur0, false, bcur,
                                                     (short)0, acc[0], false, false);
    acc[1] = __builtin_amdgcn_wmma_f32_16x16x32_bf16(false, acur1, false, bcur,
                                                     (short)0, acc[1], false, false);
    acc[2] = __builtin_amdgcn_wmma_f32_16x16x32_bf16(false, acur2, false, bcur,
                                                     (short)0, acc[2], false, false);
    acc[3] = __builtin_amdgcn_wmma_f32_16x16x32_bf16(false, acur3, false, bcur,
                                                     (short)0, acc[3], false, false);
}

// Generic fused GEMM: C[M,Ncols] = A1 @ W1^T (+ A2 @ W2^T) + bias1 (+ bias2).
// A: [M,K] bf16 row-major, W: [Ncols,K] bf16 row-major. K mult of 32, M mult of 64.
// Wave computes 64 rows x 16 cols: grid.y = M/64, (grid.x*8 + wave) = N/16 tile.
__global__ __launch_bounds__(256) void gemm_bf16_wmma(
    const __bf16* __restrict__ A1, int lda1, const __bf16* __restrict__ W1, int ldw1, int K1,
    const __bf16* __restrict__ A2, int lda2, const __bf16* __restrict__ W2, int ldw2, int K2,
    const float* __restrict__ bias1, const float* __restrict__ bias2,
    float* __restrict__ C, long long ldc)
{
    const int lane = threadIdx.x & 31;
    const int wave = threadIdx.x >> 5;
    const int m0   = blockIdx.y << 6;           // 4 M-tiles of 16 per wave
    const int n0   = (blockIdx.x * 8 + wave) << 4;
    const int r    = lane & 15;
    const int koff = (lane >> 4) << 3;          // 0 or 8: K-halves split by lane group

    vf32x8 acc[4] = {{}, {}, {}, {}};

    gemm_k_block(A1 + (size_t)(m0 + r) * lda1 + koff, lda1,
                 W1 + (size_t)(n0 + r) * ldw1 + koff, K1, acc);
    if (A2 != nullptr)
        gemm_k_block(A2 + (size_t)(m0 + r) * lda2 + koff, lda2,
                     W2 + (size_t)(n0 + r) * ldw2 + koff, K2, acc);

    const int col = n0 + r;
    float bvadd = 0.f;
    if (bias1) bvadd += bias1[col];
    if (bias2) bvadd += bias2[col];
    const int rhalf = (lane >> 4) << 3;         // C layout: lanes 16-31 hold M+8
#pragma unroll
    for (int j = 0; j < 4; ++j) {
        const int rbase = m0 + j * 16 + rhalf;
#pragma unroll
        for (int i = 0; i < 8; ++i)
            C[(size_t)(rbase + i) * (size_t)ldc + col] = acc[j][i] + bvadd;
    }
}

// Bulk f32 -> bf16 conversion, 8 elements per thread (b128 in, b128 out).
__global__ void cvt_bf16_kernel(const float* __restrict__ src, __bf16* __restrict__ dst,
                                int n8) {
    int i = blockIdx.x * 256 + threadIdx.x;
    if (i >= n8) return;
    const float4* s = reinterpret_cast<const float4*>(src) + (size_t)i * 2;
    float4 a = s[0], b = s[1];
    vbf16x8 v;
    v[0] = f2bf(a.x); v[1] = f2bf(a.y); v[2] = f2bf(a.z); v[3] = f2bf(a.w);
    v[4] = f2bf(b.x); v[5] = f2bf(b.y); v[6] = f2bf(b.z); v[7] = f2bf(b.w);
    *reinterpret_cast<vbf16x8*>(dst + (size_t)i * 8) = v;
}

// Segment start offsets from sorted batch_idx (computed once).
__global__ void seg_offsets_kernel(const int* __restrict__ bidx, int* __restrict__ seg,
                                   int N, int B) {
    int b = threadIdx.x;
    if (b > B) return;
    if (b == B) { seg[B] = N; return; }
    int lo = 0, hi = N;
    while (lo < hi) { int mid = (lo + hi) >> 1; if (bidx[mid] < b) lo = mid + 1; else hi = mid; }
    seg[b] = lo;
}

__global__ void init_state_kernel(const float* __restrict__ gf,
                                  __bf16* __restrict__ h0, __bf16* __restrict__ h1,
                                  float* __restrict__ c0, float* __restrict__ c1) {
    int i = blockIdx.x * 256 + threadIdx.x;     // B*H
    __bf16 v = f2bf(gf[i]);
    h0[i] = v; h1[i] = v; c0[i] = 0.f; c1[i] = 0.f;
}

// Segment-softmax attention: one block per batch element.
__global__ __launch_bounds__(256) void attn_kernel(
    const float* __restrict__ np, const float* __restrict__ hp,
    float* __restrict__ scores, float* __restrict__ ctx,
    const int* __restrict__ seg)
{
    const int b  = blockIdx.x;
    const int s0 = seg[b], s1 = seg[b + 1];
    const int tid = threadIdx.x;
    const int lane = tid & 31;
    const int wave = tid >> 5;
    __shared__ float red[8];
    __shared__ float mval, sval;
    const float* hb = hp + b * 512;

    // scores[n] = <node_proj[n], hp[b]> ; one wave per node
    for (int n = s0 + wave; n < s1; n += 8) {
        const float* row = np + (size_t)n * 512;
        float acc = 0.f;
        for (int k = lane; k < 512; k += 32) acc += row[k] * hb[k];
#pragma unroll
        for (int o = 16; o; o >>= 1) acc += __shfl_xor(acc, o);
        if (lane == 0) scores[n] = acc;
    }
    __threadfence_block();
    __syncthreads();

    // segment max
    float m = -3.402823466e38f;
    for (int n = s0 + tid; n < s1; n += 256) m = fmaxf(m, scores[n]);
#pragma unroll
    for (int o = 16; o; o >>= 1) m = fmaxf(m, __shfl_xor(m, o));
    if (lane == 0) red[wave] = m;
    __syncthreads();
    if (tid == 0) {
        float mm = red[0];
#pragma unroll
        for (int i = 1; i < 8; ++i) mm = fmaxf(mm, red[i]);
        mval = mm;
    }
    __syncthreads();
    m = mval;

    // segment sum of exp
    float s = 0.f;
    for (int n = s0 + tid; n < s1; n += 256) s += __expf(scores[n] - m);
#pragma unroll
    for (int o = 16; o; o >>= 1) s += __shfl_xor(s, o);
    if (lane == 0) red[wave] = s;
    __syncthreads();
    if (tid == 0) {
        float ss = 0.f;
#pragma unroll
        for (int i = 0; i < 8; ++i) ss += red[i];
        sval = ss;
    }
    __syncthreads();
    const float inv = 1.f / sval;

    // normalized weights in place
    for (int n = s0 + tid; n < s1; n += 256) scores[n] = __expf(scores[n] - m) * inv;
    __threadfence_block();
    __syncthreads();

    // ctx[b] = sum_n w[n] * node_proj[n]
    for (int h = tid; h < 512; h += 256) {
        float acc = 0.f;
        for (int n = s0; n < s1; ++n) acc += scores[n] * np[(size_t)n * 512 + h];
        ctx[b * 512 + h] = acc;
    }
}

// x_t = concat(embedding[captions[:, t]], ctx) -> bf16
__global__ void build_x_kernel(const float* __restrict__ emb, const int* __restrict__ cap,
                               const float* __restrict__ ctx, __bf16* __restrict__ x,
                               int t, int T) {
    int i = blockIdx.x * 256 + threadIdx.x;     // B * 1024
    int b = i >> 10, j = i & 1023;
    float v;
    if (j < 512) { int tok = cap[b * T + t]; v = emb[(size_t)tok * 512 + j]; }
    else          v = ctx[b * 512 + (j - 512)];
    x[i] = f2bf(v);
}

// LSTM cell nonlinearity (PyTorch gate order i,f,g,o); gates include both biases.
// Writes h directly in bf16 (only ever consumed as a WMMA A-operand).
__global__ void lstm_cell_kernel(const float* __restrict__ gates,
                                 const float* __restrict__ c_in,
                                 __bf16* __restrict__ h_out, float* __restrict__ c_out) {
    int idx = blockIdx.x * 256 + threadIdx.x;   // B*H
    int b = idx >> 9, h = idx & 511;
    const float* g = gates + (size_t)b * 2048;
    float gi = g[h], gf = g[512 + h], gg = g[1024 + h], go = g[1536 + h];
    float si = 1.f / (1.f + __expf(-gi));
    float sf = 1.f / (1.f + __expf(-gf));
    float so = 1.f / (1.f + __expf(-go));
    float c2 = sf * c_in[idx] + si * tanhf(gg);
    c_out[idx] = c2;
    h_out[idx] = f2bf(so * tanhf(c2));
}

extern "C" void kernel_launch(void* const* d_in, const int* in_sizes, int n_in,
                              void* d_out, int out_size, void* d_ws, size_t ws_size,
                              hipStream_t stream) {
    (void)in_sizes; (void)n_in; (void)out_size; (void)ws_size;
    const float* graph   = (const float*)d_in[0];
    const float* nodef   = (const float*)d_in[1];
    const float* emb     = (const float*)d_in[2];
    const float* W_a     = (const float*)d_in[3];
    const float* b_a     = (const float*)d_in[4];
    const float* W_c     = (const float*)d_in[5];
    const float* b_c     = (const float*)d_in[6];
    const float* W_ih0   = (const float*)d_in[7];
    const float* W_hh0   = (const float*)d_in[8];
    const float* b_ih0   = (const float*)d_in[9];
    const float* b_hh0   = (const float*)d_in[10];
    const float* W_ih1   = (const float*)d_in[11];
    const float* W_hh1   = (const float*)d_in[12];
    const float* b_ih1   = (const float*)d_in[13];
    const float* b_hh1   = (const float*)d_in[14];
    const float* W_fc    = (const float*)d_in[15];
    const float* b_fc    = (const float*)d_in[16];
    const int*   bidx    = (const int*)d_in[17];
    const int*   caps    = (const int*)d_in[18];
    float* out = (float*)d_out;

    constexpr int B = 64, T = 20, H = 512, E = 512, V = 32000, N = 8192;

    // workspace carve-out (~70 MB)
    char* w = (char*)d_ws;
    size_t off = 0;
    auto take = [&](size_t bytes) -> void* {
        void* p = w + off; off += (bytes + 255) & ~(size_t)255; return p;
    };
    float*  np      = (float*)take((size_t)N * H * 4);          // node_proj (f32)
    __bf16* nodefb  = (__bf16*)take((size_t)N * H * 2);
    __bf16* Wab     = (__bf16*)take((size_t)H * H * 2);
    __bf16* Wcb     = (__bf16*)take((size_t)H * H * 2);
    __bf16* Wih0b   = (__bf16*)take((size_t)4 * H * (E + H) * 2);
    __bf16* Whh0b   = (__bf16*)take((size_t)4 * H * H * 2);
    __bf16* Wih1b   = (__bf16*)take((size_t)4 * H * H * 2);
    __bf16* Whh1b   = (__bf16*)take((size_t)4 * H * H * 2);
    __bf16* Wfcb    = (__bf16*)take((size_t)V * H * 2);
    float*  hp      = (float*)take((size_t)B * H * 4);
    float*  ctx     = (float*)take((size_t)B * H * 4);
    __bf16* xbuf    = (__bf16*)take((size_t)B * (E + H) * 2);
    float*  gates   = (float*)take((size_t)B * 4 * H * 4);
    __bf16* h0      = (__bf16*)take((size_t)B * H * 2);
    __bf16* h1      = (__bf16*)take((size_t)B * H * 2);
    float*  c0      = (float*)take((size_t)B * H * 4);
    float*  c1      = (float*)take((size_t)B * H * 4);
    float*  sc      = (float*)take((size_t)N * 4);
    int*    seg     = (int*)take((size_t)(B + 1) * 4);

    auto cvt = [&](const float* s, __bf16* d, size_t n) {
        int n8 = (int)(n / 8);
        cvt_bf16_kernel<<<(n8 + 255) / 256, 256, 0, stream>>>(s, d, n8);
    };

    // one-time work: weight/feature conversion, offsets, state init, node_proj
    cvt(nodef, nodefb, (size_t)N * H);
    cvt(W_a,   Wab,   (size_t)H * H);
    cvt(W_c,   Wcb,   (size_t)H * H);
    cvt(W_ih0, Wih0b, (size_t)4 * H * (E + H));
    cvt(W_hh0, Whh0b, (size_t)4 * H * H);
    cvt(W_ih1, Wih1b, (size_t)4 * H * H);
    cvt(W_hh1, Whh1b, (size_t)4 * H * H);
    cvt(W_fc,  Wfcb,  (size_t)V * H);

    seg_offsets_kernel<<<1, 128, 0, stream>>>(bidx, seg, N, B);
    init_state_kernel<<<(B * H) / 256, 256, 0, stream>>>(graph, h0, h1, c0, c1);
    // node_proj = node_features @ W_c^T + b_c   (8192x512 @ 512x512)
    gemm_bf16_wmma<<<dim3(H / 128, N / 64), 256, 0, stream>>>(
        nodefb, H, Wcb, H, H, nullptr, 0, nullptr, 0, 0, b_c, nullptr, np, H);

    for (int t = 0; t < T; ++t) {
        // hp = h1 @ W_a^T + b_a
        gemm_bf16_wmma<<<dim3(H / 128, B / 64), 256, 0, stream>>>(
            h1, H, Wab, H, H, nullptr, 0, nullptr, 0, 0, b_a, nullptr, hp, H);
        // segment softmax + context
        attn_kernel<<<B, 256, 0, stream>>>(np, hp, sc, ctx, seg);
        // x_t = [emb_t | ctx]
        build_x_kernel<<<(B * (E + H)) / 256, 256, 0, stream>>>(emb, caps, ctx, xbuf, t, T);
        // layer-0 gates = x @ W_ih0^T + h0 @ W_hh0^T + b_ih0 + b_hh0
        gemm_bf16_wmma<<<dim3((4 * H) / 128, B / 64), 256, 0, stream>>>(
            xbuf, E + H, Wih0b, E + H, E + H,
            h0, H, Whh0b, H, H,
            b_ih0, b_hh0, gates, 4 * H);
        lstm_cell_kernel<<<(B * H) / 256, 256, 0, stream>>>(gates, c0, h0, c0);
        // layer-1 gates = h0n @ W_ih1^T + h1 @ W_hh1^T + b_ih1 + b_hh1
        gemm_bf16_wmma<<<dim3((4 * H) / 128, B / 64), 256, 0, stream>>>(
            h0, H, Wih1b, H, H,
            h1, H, Whh1b, H, H,
            b_ih1, b_hh1, gates, 4 * H);
        lstm_cell_kernel<<<(B * H) / 256, 256, 0, stream>>>(gates, c1, h1, c1);
        // logits -> out[:, t, :]  (64x512 @ 512x32000)
        gemm_bf16_wmma<<<dim3(V / 128, B / 64), 256, 0, stream>>>(
            h1, H, Wfcb, H, H, nullptr, 0, nullptr, 0, 0,
            b_fc, nullptr, out + (size_t)t * V, (long long)T * V);
    }
}
